// ANEOptimizedAttention_23897198035757
// MI455X (gfx1250) — matrix operand order repaired and checked by reading, not compiled, not run on hardware
//
#include <hip/hip_runtime.h>

// ---------------------------------------------------------------------------
// Fused MHA for B=2, C=1024, H=16, S=2048, D=64 on gfx1250 (MI455X).
// Pipeline:
//   1) proj_gemm (WMMA bf16): Q = Wq*x  -> qT  [B][H][S][D] bf16 (transposed)
//   2) proj_gemm            : K = Wk*x  -> kT  [B][H][S][D] bf16 (transposed)
//   3) proj_gemm            : V = Wv*x  -> v   [B][C][S]    bf16 (row-major)
//   4) flash_attn (WMMA bf16, online softmax, async double-buffered K/V
//      staging via GLOBAL_LOAD_ASYNC_TO_LDS_B128) -> o [B][C][S] fp32
//   5) proj_gemm            : out = Wo*o + bo -> d_out fp32
// ---------------------------------------------------------------------------

#define DEVINL __device__ __forceinline__

typedef __bf16 bf16_t;
typedef __attribute__((ext_vector_type(16))) __bf16        bf16x16;
typedef __attribute__((ext_vector_type(8)))  float         f32x8;
typedef __attribute__((ext_vector_type(4)))  float         f32x4;
typedef __attribute__((ext_vector_type(4)))  unsigned int  u32x4;

constexpr int B_ = 2, C_ = 1024, H_ = 16, S_ = 2048, D_ = 64;
constexpr float SCALE_ = 0.125f;  // D^-0.5

enum { OUT_F32 = 0, OUT_BF16 = 1, OUT_BF16_HT = 2 };

// fp32 -> bf16 round-to-nearest-even (pure integer ops)
DEVINL bf16_t f2bf(float f) {
  unsigned u = __builtin_bit_cast(unsigned, f);
  u += 0x7FFFu + ((u >> 16) & 1u);
  unsigned short h = (unsigned short)(u >> 16);
  return __builtin_bit_cast(bf16_t, h);
}

// Async 16B global -> LDS copy (CDNA5 GLOBAL_LOAD_ASYNC_TO_LDS_B128, ASYNCcnt).
// LDS destination address = low 32 bits of the generic pointer (ISA 10.2: LDS
// aperture maps addr[31:0] to the LDS offset).
DEVINL void async_copy_b128(void* lds_dst, const void* gsrc) {
  unsigned lds = (unsigned)(uintptr_t)lds_dst;
  asm volatile("global_load_async_to_lds_b128 %0, %1, off"
               :: "v"(lds), "v"(gsrc)
               : "memory");
}
DEVINL void wait_asynccnt0() {
  asm volatile("s_wait_asynccnt 0x0" ::: "memory");
}

// A-matrix fragment (16x32 bf16, M x K). Per ISA 7.12.2:
//   lane m = lane&15; lanes 0-15 hold K 0..7 / 16..23, lanes 16-31 hold K 8..15 / 24..31.
DEVINL bf16x16 load_frag_a(const bf16_t* rowBase, int rowStride, int lane) {
  const int half = lane >> 4, m = lane & 15;
  const bf16_t* p = rowBase + m * rowStride + half * 8;
  union { u32x4 u[2]; bf16x16 v; } t;
  t.u[0] = *(const u32x4*)(p);
  t.u[1] = *(const u32x4*)(p + 16);
  return t.v;
}

// B-matrix fragment (32x16 bf16, K x N) from a [n][K-contiguous] layout. Per ISA:
//   lane n = lane&15; lanes 0-15 hold K 0..7 / 16..23, lanes 16-31 hold K 8..15 / 24..31.
DEVINL bf16x16 load_frag_b(const bf16_t* colBase, int colStride, int lane) {
  const int half = lane >> 4, n = lane & 15;
  const bf16_t* p = colBase + n * colStride + half * 8;
  union { u32x4 u[2]; bf16x16 v; } t;
  t.u[0] = *(const u32x4*)(p);
  t.u[1] = *(const u32x4*)(p + 16);
  return t.v;
}

DEVINL f32x8 wmma_bf16(bf16x16 a, bf16x16 b, f32x8 c) {
  return __builtin_amdgcn_wmma_f32_16x16x32_bf16(
      /*neg_a=*/false, a, /*neg_b=*/false, b,
      /*c_mod=*/(short)0, c, /*reuse_a=*/false, /*reuse_b=*/false);
}

// ---------------------------------------------------------------------------
// Tiled GEMM: Y[m][n] = sum_k A[m][k] * X[k][n] + bias[m]
// M = K = C_ = 1024, N = S_ = 2048. 128x128 tile per block, 8 waves (4M x 2N),
// each wave: 2x4 accumulator tiles, K-step 32, fp32->bf16 convert into LDS.
// ---------------------------------------------------------------------------
__global__ __launch_bounds__(256)
void proj_gemm_kernel(const float* __restrict__ A,     // [C_][C_] weights
                      const float* __restrict__ Xall,  // [B_][C_][S_]
                      const float* __restrict__ bias,  // [C_]
                      void* __restrict__ outAll, int mode) {
  constexpr int M = C_, N = S_, K = C_;
  __shared__ bf16_t As[128 * 32];  // [m][k] row-major
  __shared__ bf16_t Bs[128 * 32];  // [n][k] (K contiguous per column => B-frag friendly)

  const int tid  = threadIdx.x;
  const int lane = tid & 31;
  const int wv   = tid >> 5;
  const int half = lane >> 4;

  const int nBlock = blockIdx.x * 128;
  const int mBlock = blockIdx.y * 128;
  const int batch  = blockIdx.z;
  const float* X = Xall + (size_t)batch * K * N;

  f32x8 acc[2][4] = {};

  const int wm = (wv >> 1) * 32;  // wave M offset within tile
  const int wn = (wv & 1) * 64;   // wave N offset within tile

  const int ar = tid >> 3, ac = (tid & 7) * 4;  // A staging coords
  const int br = tid & 31, bc = tid >> 5;       // B staging coords

  for (int k0 = 0; k0 < K; k0 += 32) {
    __syncthreads();
    // Stage A tile: 128x32 fp32 -> bf16
#pragma unroll
    for (int i = 0; i < 4; ++i) {
      const int row = ar + i * 32;
      f32x4 v = *(const f32x4*)(A + (size_t)(mBlock + row) * K + k0 + ac);
#pragma unroll
      for (int j = 0; j < 4; ++j) As[row * 32 + ac + j] = f2bf(v[j]);
    }
    // Stage B tile: rows k0..k0+32, cols nBlock..+128, transposed into [n][k]
#pragma unroll
    for (int i = 0; i < 4; ++i) {
      const int nc = (bc + i * 8) * 4;
      f32x4 v = *(const f32x4*)(X + (size_t)(k0 + br) * N + nBlock + nc);
#pragma unroll
      for (int j = 0; j < 4; ++j) Bs[(nc + j) * 32 + br] = f2bf(v[j]);
    }
    if (k0 + 32 < K) {  // prefetch next tiles -> global_prefetch_b8
      __builtin_prefetch(A + (size_t)(mBlock + ar) * K + k0 + 32 + ac, 0, 1);
      __builtin_prefetch(X + (size_t)(k0 + 32 + br) * N + nBlock + bc * 4, 0, 1);
    }
    __syncthreads();

    bf16x16 af[2], bfr[4];
#pragma unroll
    for (int i = 0; i < 2; ++i)
      af[i] = load_frag_a(&As[(wm + i * 16) * 32], 32, lane);
#pragma unroll
    for (int j = 0; j < 4; ++j)
      bfr[j] = load_frag_b(&Bs[(wn + j * 16) * 32], 32, lane);
#pragma unroll
    for (int i = 0; i < 2; ++i)
#pragma unroll
      for (int j = 0; j < 4; ++j)
        acc[i][j] = wmma_bf16(af[i], bfr[j], acc[i][j]);
  }

  // Epilogue. C/D layout: m = r + 8*(lane>=16), n = lane&15 per 16x16 tile.
#pragma unroll
  for (int i = 0; i < 2; ++i) {
#pragma unroll
    for (int j = 0; j < 4; ++j) {
#pragma unroll
      for (int r = 0; r < 8; ++r) {
        const int m = mBlock + wm + i * 16 + r + 8 * half;
        const int n = nBlock + wn + j * 16 + (lane & 15);
        const float val = acc[i][j][r] + bias[m];
        if (mode == OUT_F32) {
          ((float*)outAll)[(size_t)batch * M * N + (size_t)m * N + n] = val;
        } else if (mode == OUT_BF16) {
          ((bf16_t*)outAll)[(size_t)batch * M * N + (size_t)m * N + n] = f2bf(val);
        } else {  // OUT_BF16_HT: [B][H][S][D] (per-head transposed)
          const int h = m >> 6, d = m & 63;
          ((bf16_t*)outAll)[(size_t)batch * M * N + ((size_t)h * N + n) * 64 + d] =
              f2bf(val);
        }
      }
    }
  }
}

// ---------------------------------------------------------------------------
// Flash attention. Grid (S/128, H, B), 256 threads (8 waves). Wave w owns 16
// target columns t in [blk*128 + 16w, +16). Q B-fragments live in registers
// across the whole s-loop. K^T / V tiles are double-buffered in LDS and filled
// with GLOBAL_LOAD_ASYNC_TO_LDS_B128: tiles for step i+1 are issued right
// after the barrier, overlapping the WMMA work of step i; each wave drains its
// own ASYNCcnt before the barrier that publishes the buffer.
// ---------------------------------------------------------------------------
__global__ __launch_bounds__(256)
void flash_attn_kernel(const bf16_t* __restrict__ qT,  // [B][H][S][D]
                       const bf16_t* __restrict__ kT,  // [B][H][S][D]
                       const bf16_t* __restrict__ vP,  // [B][C][S]
                       float* __restrict__ oP) {       // [B][C][S] fp32
  __shared__ bf16_t kTile[2][32 * 64];     // [s][d]   (A-frag layout for scores)
  __shared__ bf16_t vTile[2][64 * 32];     // [d][s]   (A-frag layout for output)
  __shared__ bf16_t pTile[8 * 16 * 32];    // per-wave [t][s] (B-frag layout)

  const int tid  = threadIdx.x;
  const int lane = tid & 31;
  const int wv   = tid >> 5;
  const int half = lane >> 4;
  const int tl   = lane & 15;

  const int h = blockIdx.y;
  const int b = blockIdx.z;
  const int tBase = blockIdx.x * 128 + wv * 16;

  const bf16_t* qTh = qT + ((size_t)b * H_ + h) * (size_t)S_ * D_;
  const bf16_t* kTh = kT + ((size_t)b * H_ + h) * (size_t)S_ * D_;
  const bf16_t* vh  = vP + ((size_t)b * C_ + (size_t)h * D_) * S_;
  float*        oh  = oP + ((size_t)b * C_ + (size_t)h * D_) * S_;

  // Q fragments for this wave's 16 columns (d-steps 0..31 and 32..63).
  bf16x16 qb[2];
#pragma unroll
  for (int j = 0; j < 2; ++j)
    qb[j] = load_frag_b(qTh + (size_t)tBase * D_ + j * 32, D_, lane);

  f32x8 oacc[4] = {};                 // O tiles: d = 16*dt + r + 8*half, n = tl
  float m_run = -1e30f, l_run = 0.0f;

  const int kr = tid >> 3, kc = (tid & 7) * 8;  // kTile staging (32x64)
  const int vr = tid >> 2, vc = (tid & 3) * 8;  // vTile staging (64x32)

  // Issue one 16B async copy per thread per tile (2 per thread total per step).
  auto issue_tiles = [&](int buf, int s0) {
    async_copy_b128(&kTile[buf][kr * 64 + kc],
                    kTh + (size_t)(s0 + kr) * D_ + kc);
    async_copy_b128(&vTile[buf][vr * 32 + vc],
                    vh + (size_t)vr * S_ + s0 + vc);
  };

  issue_tiles(0, 0);  // prologue: fill buffer 0

  for (int s0 = 0; s0 < S_; s0 += 32) {
    const int buf = (s0 >> 5) & 1;
    wait_asynccnt0();    // my async writes into `buf` (and any in-flight) done
    __syncthreads();     // publish all waves' writes; back buffer reads retired
    if (s0 + 32 < S_) issue_tiles(buf ^ 1, s0 + 32);  // overlap with compute

    // Scores: sc[si][t] for s-subtiles si (16 rows), contraction over d (2 steps).
    f32x8 sc[2] = {};
#pragma unroll
    for (int si = 0; si < 2; ++si)
#pragma unroll
      for (int j = 0; j < 2; ++j) {
        bf16x16 ak = load_frag_a(&kTile[buf][(si * 16) * 64 + j * 32], 64, lane);
        sc[si] = wmma_bf16(ak, qb[j], sc[si]);
      }

    // Online softmax over s for each target column t (lane-half pair = column).
    float mx = -1e30f;
#pragma unroll
    for (int si = 0; si < 2; ++si)
#pragma unroll
      for (int r = 0; r < 8; ++r) {
        sc[si][r] *= SCALE_;
        mx = fmaxf(mx, sc[si][r]);
      }
    mx = fmaxf(mx, __shfl_xor(mx, 16, 32));
    const float m_new = fmaxf(m_run, mx);
    const float corr  = __expf(m_run - m_new);
    float ssum = 0.0f;
#pragma unroll
    for (int si = 0; si < 2; ++si)
#pragma unroll
      for (int r = 0; r < 8; ++r) {
        const float p = __expf(sc[si][r] - m_new);
        sc[si][r] = p;
        ssum += p;
      }
    ssum += __shfl_xor(ssum, 16, 32);
    l_run = l_run * corr + ssum;
    m_run = m_new;

    // Write P^T (bf16) to wave-private LDS in [t][s] B-fragment layout.
    bf16_t* pw = &pTile[wv * 16 * 32];
#pragma unroll
    for (int si = 0; si < 2; ++si)
#pragma unroll
      for (int r = 0; r < 8; ++r)
        pw[tl * 32 + si * 16 + r + 8 * half] = f2bf(sc[si][r]);

    // Rescale accumulators (each lane's column t is fixed => scalar corr works).
#pragma unroll
    for (int dt = 0; dt < 4; ++dt)
#pragma unroll
      for (int r = 0; r < 8; ++r) oacc[dt][r] *= corr;

    // O += V[d][s] * P[s][t]   (K = 32 source rows in one WMMA per d-tile)
    bf16x16 pb = load_frag_b(pw, 32, lane);
#pragma unroll
    for (int dt = 0; dt < 4; ++dt) {
      bf16x16 av = load_frag_a(&vTile[buf][(dt * 16) * 32], 32, lane);
      oacc[dt] = wmma_bf16(av, pb, oacc[dt]);
    }
  }

  const float inv = 1.0f / l_run;
#pragma unroll
  for (int dt = 0; dt < 4; ++dt)
#pragma unroll
    for (int r = 0; r < 8; ++r) {
      const int d = dt * 16 + r + 8 * half;
      oh[(size_t)d * S_ + tBase + tl] = oacc[dt][r] * inv;
    }
}

// ---------------------------------------------------------------------------
extern "C" void kernel_launch(void* const* d_in, const int* in_sizes, int n_in,
                              void* d_out, int out_size, void* d_ws, size_t ws_size,
                              hipStream_t stream) {
  const float* x_q = (const float*)d_in[0];
  const float* x_k = (const float*)d_in[1];
  const float* x_v = (const float*)d_in[2];
  const float* Wq  = (const float*)d_in[3];
  const float* bq  = (const float*)d_in[4];
  const float* Wk  = (const float*)d_in[5];
  const float* bk  = (const float*)d_in[6];
  const float* Wv  = (const float*)d_in[7];
  const float* bv  = (const float*)d_in[8];
  const float* Wo  = (const float*)d_in[9];
  const float* bo  = (const float*)d_in[10];

  const size_t nBCS = (size_t)B_ * C_ * S_;  // 4,194,304 elements
  // Workspace layout (~42 MB): qT, kT, v (bf16) then o (fp32).
  bf16_t* qT = (bf16_t*)d_ws;
  bf16_t* kT = qT + nBCS;
  bf16_t* vP = kT + nBCS;
  float*  oP = (float*)(vP + nBCS);

  const dim3 gProj(S_ / 128, C_ / 128, B_);  // (16, 8, 2)
  const dim3 gAttn(S_ / 128, H_, B_);        // (16, 16, 2)

  proj_gemm_kernel<<<gProj, 256, 0, stream>>>(Wq, x_q, bq, qT, OUT_BF16_HT);
  proj_gemm_kernel<<<gProj, 256, 0, stream>>>(Wk, x_k, bk, kT, OUT_BF16_HT);
  proj_gemm_kernel<<<gProj, 256, 0, stream>>>(Wv, x_v, bv, vP, OUT_BF16);
  flash_attn_kernel<<<gAttn, 256, 0, stream>>>(qT, kT, vP, oP);
  proj_gemm_kernel<<<gProj, 256, 0, stream>>>(Wo, oP, bo, d_out, OUT_F32);
}
